// GAT_23141283790914
// MI455X (gfx1250) — matrix operand order repaired and checked by reading, not compiled, not run on hardware
//
#include <hip/hip_runtime.h>
#include <hip/hip_bf16.h>

typedef _Float16 half_t;
typedef __attribute__((ext_vector_type(16))) _Float16 v16h;
typedef __attribute__((ext_vector_type(8)))  float    v8f;

#define GN 50000
#define GNP 50048            // 782 * 64: row-padded so GEMM needs no bounds checks
#define GE 800000
#define GIN 128
#define GHID 16
#define GHEADS 8
#define GCLS 40

// ---------- small utility kernels ----------

__global__ void fill_f32_kernel(float* p, float v, long long n) {
  long long i = (long long)blockIdx.x * blockDim.x + threadIdx.x;
  if (i < n) p[i] = v;
}

__global__ void fill_u32_kernel(unsigned* p, unsigned v, long long n) {
  long long i = (long long)blockIdx.x * blockDim.x + threadIdx.x;
  if (i < n) p[i] = v;
}

__global__ void f32_to_f16_kernel(const float* __restrict__ src, half_t* __restrict__ dst, long long n) {
  long long i = (long long)blockIdx.x * blockDim.x + threadIdx.x;
  if (i < n) dst[i] = (half_t)src[i];
}

// W1 [128,128] f32 row-major -> f16 column-major Bt[n*128 + k] = W1[k,n]
__global__ void pack_w1t_kernel(const float* __restrict__ W1, half_t* __restrict__ dst) {
  int i = blockIdx.x * blockDim.x + threadIdx.x;
  if (i < 128 * 128) {
    int n = i / 128, k = i % 128;
    dst[i] = (half_t)W1[k * 128 + n];
  }
}

// W2 [128,40] f32 row-major -> f16 column-major padded to 48 cols: Bt[n*128 + k]
__global__ void pack_w2t_kernel(const float* __restrict__ W2, half_t* __restrict__ dst) {
  int i = blockIdx.x * blockDim.x + threadIdx.x;
  if (i < 48 * 128) {
    int n = i / 128, k = i % 128;
    dst[i] = (n < GCLS) ? (half_t)W2[k * GCLS + n] : (half_t)0.f;
  }
}

// ---------- WMMA GEMM: C[GNP,NC] = A[GNP,128] * B[128,NC] ----------
// A is row-padded to GNP rows (pad tiles compute garbage into pad rows that are
// never read; N % 16 == 0 so valid rows never share a tile with pad rows).
// Bt is column-major so each lane's B fragment is 32 contiguous LDS bytes.
// 256 threads = 8 waves; 4 row-tiles per block; waves stride over
// (row-tile x col-tile) items; `wave` goes through readfirstlane so control
// flow is scalar and EXEC is provably all-1s around every WMMA.

template <int NC>
__global__ __launch_bounds__(256) void gemm_wmma_kernel(const half_t* __restrict__ A,
                                                        const half_t* __restrict__ Bt,
                                                        float* __restrict__ C) {
  constexpr int K  = 128;
  constexpr int RT = 4;          // row tiles per block (64 rows)
  constexpr int CT = NC / 16;    // col tiles
  __shared__ __attribute__((aligned(16))) half_t sA[RT * 16 * K];   // 16 KB
  __shared__ __attribute__((aligned(16))) half_t sBt[NC * K];       // col-major weights

  const int tid  = threadIdx.x;
  const int wave = __builtin_amdgcn_readfirstlane(tid >> 5);
  const int lane = tid & 31;
  const int m0   = blockIdx.x * (RT * 16);

  // vectorized staging: A tile is a flat contiguous region (rows contiguous)
  {
    const uint4* gA = (const uint4*)(A + (size_t)m0 * K);
    uint4* s4 = (uint4*)sA;
#pragma unroll
    for (int i = 0; i < (RT * 16 * K) / 8 / 256; ++i) s4[tid + i * 256] = gA[tid + i * 256];
    const uint4* gB = (const uint4*)Bt;
    uint4* sb4 = (uint4*)sBt;
    for (int i = tid; i < (NC * K) / 8; i += 256) sb4[i] = gB[i];
  }
  __syncthreads();

  const int hi = lane >> 4;   // 0: lanes 0-15, 1: lanes 16-31
  const int l  = lane & 15;

  for (int t = wave; t < RT * CT; t += 8) {
    const int rt = t / CT, ct = t % CT;
    const half_t* arow = &sA[(rt * 16 + l) * K];
    const half_t* bcol = &sBt[(ct * 16 + l) * K];
    v8f acc = {};
#pragma unroll
    for (int kk = 0; kk < K; kk += 32) {
      v16h av, bv;
      // A 16x32 f16: lanes 0-15 hold K 0..15, lanes 16-31 hold K 16..31 (row = rt*16+l)
#pragma unroll
      for (int i = 0; i < 16; ++i) av[i] = arow[kk + hi * 16 + i];
      // B 32x16 f16: lanes 0-15 hold K 0..15, lanes 16-31 hold K 16..31 (col = ct*16+l)
#pragma unroll
      for (int i = 0; i < 16; ++i) bv[i] = bcol[kk + hi * 16 + i];
      acc = __builtin_amdgcn_wmma_f32_16x16x32_f16(false, av, false, bv,
                                                   (short)0, acc, false, false);
    }
    // C/D: col = ct*16 + lane%16 ; vgpr r -> row (lane>=16 ? 8 : 0) + r
    float* crow = C + (size_t)(m0 + rt * 16 + hi * 8) * NC + ct * 16 + l;
#pragma unroll
    for (int r = 0; r < 8; ++r) crow[(size_t)r * NC] = acc[r];
  }
}

// ---------- attention logits ----------

template <int D>
__global__ void el_er_kernel(const float* __restrict__ feat, const float* __restrict__ al,
                             const float* __restrict__ ar, float* __restrict__ el,
                             float* __restrict__ er, int n_nodes, int H, int ldF) {
  int i = blockIdx.x * blockDim.x + threadIdx.x;
  if (i >= n_nodes * H) return;
  int n = i / H, h = i % H;
  const float* f = feat + (size_t)n * ldF + h * D;
  float sl = 0.f, sr = 0.f;
#pragma unroll
  for (int d = 0; d < D; ++d) {
    float v = f[d];
    sl += v * al[h * D + d];
    sr += v * ar[h * D + d];
  }
  el[i] = sl;
  er[i] = sr;
}

// ---------- segment softmax helpers ----------

__device__ __forceinline__ float leaky(float x) { return x > 0.f ? x : 0.2f * x; }

__device__ __forceinline__ unsigned fkey(float f) {
  unsigned u = __float_as_uint(f);
  return (u & 0x80000000u) ? ~u : (u | 0x80000000u);
}
__device__ __forceinline__ float fdecode(unsigned k) {
  unsigned u = (k & 0x80000000u) ? (k & 0x7FFFFFFFu) : ~k;
  return __uint_as_float(u);
}
__device__ __forceinline__ float clamp_finite(float m) {
  return (fabsf(m) < 3.0e38f) ? m : 0.f;   // NaN/inf -> 0, matches reference
}

__global__ void edge_max_kernel(const int* __restrict__ src, const int* __restrict__ dst,
                                const float* __restrict__ el, const float* __restrict__ er,
                                unsigned* __restrict__ emaxk, int nE, int H) {
  int i = blockIdx.x * blockDim.x + threadIdx.x;
  if (i >= nE * H) return;
  int e = i / H, h = i % H;
  int s = src[e], d = dst[e];
  float v = leaky(el[s * H + h] + er[d * H + h]);
  atomicMax(&emaxk[d * H + h], fkey(v));
}

__global__ void edge_denom_kernel(const int* __restrict__ src, const int* __restrict__ dst,
                                  const float* __restrict__ el, const float* __restrict__ er,
                                  const unsigned* __restrict__ emaxk, float* __restrict__ denom,
                                  int nE, int H) {
  int i = blockIdx.x * blockDim.x + threadIdx.x;
  if (i >= nE * H) return;
  int e = i / H, h = i % H;
  int s = src[e], d = dst[e];
  float m  = clamp_finite(fdecode(emaxk[d * H + h]));
  float ee = expf(leaky(el[s * H + h] + er[d * H + h]) - m);
  atomicAdd(&denom[d * H + h], ee);
}

template <int D>
__global__ void edge_aggr_kernel(const int* __restrict__ src, const int* __restrict__ dst,
                                 const float* __restrict__ el, const float* __restrict__ er,
                                 const unsigned* __restrict__ emaxk, const float* __restrict__ denom,
                                 const float* __restrict__ feat, float* __restrict__ out,
                                 int nE, int H, int ldF, int ldO) {
  int i = blockIdx.x * blockDim.x + threadIdx.x;
  if (i >= nE * H) return;
  int e = i / H, h = i % H;
  int s = src[e], d = dst[e];
  float m     = clamp_finite(fdecode(emaxk[d * H + h]));
  float ee    = expf(leaky(el[s * H + h] + er[d * H + h]) - m);
  float alpha = ee / (denom[d * H + h] + 1e-9f);
  const float* fs = feat + (size_t)s * ldF + h * D;
  float* od       = out  + (size_t)d * ldO + h * D;
#pragma unroll
  for (int k = 0; k < D; ++k) atomicAdd(&od[k], alpha * fs[k]);
}

// ---------- layer finalizers ----------

__global__ void finalize1_kernel(const float* __restrict__ out1, const float* __restrict__ b1,
                                 half_t* __restrict__ hh, long long total) {
  long long i = (long long)blockIdx.x * blockDim.x + threadIdx.x;
  if (i >= total) return;
  int c = (int)(i & 127);
  float x = out1[i] + b1[c];
  float y = x > 0.f ? x : (expf(x) - 1.f);
  hh[i] = (half_t)y;
}

__global__ void add_bias_kernel(float* __restrict__ out, const float* __restrict__ b,
                                long long total, int C) {
  long long i = (long long)blockIdx.x * blockDim.x + threadIdx.x;
  if (i >= total) return;
  out[i] += b[i % C];
}

// ---------- orchestration ----------

static inline int nblk(long long n, int t) { return (int)((n + t - 1) / t); }

extern "C" void kernel_launch(void* const* d_in, const int* in_sizes, int n_in,
                              void* d_out, int out_size, void* d_ws, size_t ws_size,
                              hipStream_t stream) {
  const float* features = (const float*)d_in[0];
  const int*   esrc     = (const int*)d_in[1];
  const int*   edst     = (const int*)d_in[2];
  const float* W1       = (const float*)d_in[3];
  const float* al1      = (const float*)d_in[4];
  const float* ar1      = (const float*)d_in[5];
  const float* b1       = (const float*)d_in[6];
  const float* W2       = (const float*)d_in[7];
  const float* al2      = (const float*)d_in[8];
  const float* ar2      = (const float*)d_in[9];
  const float* b2       = (const float*)d_in[10];
  float* out = (float*)d_out;

  char* ws = (char*)d_ws;
  size_t off = 0;
  auto carve = [&](size_t bytes) -> char* {
    char* p = ws + off;
    off = (off + bytes + 255) & ~(size_t)255;
    return p;
  };
  // row-padded (GNP) buffers for the GEMM path
  half_t*   Ah    = (half_t*)carve((size_t)GNP * GIN * 2);  // features f16 (+pad rows)
  half_t*   W1t   = (half_t*)carve((size_t)128 * 128 * 2);  // W1 f16 col-major
  float*    feat1 = (float*)carve((size_t)GNP * 128 * 4);   // X @ W1
  float*    el1   = (float*)carve((size_t)GN * GHEADS * 4);
  float*    er1   = (float*)carve((size_t)GN * GHEADS * 4);
  unsigned* emax1 = (unsigned*)carve((size_t)GN * GHEADS * 4);
  float*    den1  = (float*)carve((size_t)GN * GHEADS * 4);
  float*    out1  = (float*)carve((size_t)GN * 128 * 4);    // aggregated layer-1
  half_t*   hh    = (half_t*)carve((size_t)GNP * 128 * 2);  // elu(out1+b1) f16 (+pad rows)
  half_t*   W2t   = (half_t*)carve((size_t)48 * 128 * 2);   // W2 f16 col-major padded
  float*    feat2 = (float*)carve((size_t)GNP * 48 * 4);    // h @ W2 (ld 48)
  float*    el2   = (float*)carve((size_t)GN * 4);
  float*    er2   = (float*)carve((size_t)GN * 4);
  unsigned* emax2 = (unsigned*)carve((size_t)GN * 4);
  float*    den2  = (float*)carve((size_t)GN * 4);
  (void)ws_size; (void)in_sizes; (void)n_in; (void)out_size;

  const int T = 256;

  // ---- precision conversion / weight transpose ----
  f32_to_f16_kernel<<<nblk((long long)GN * GIN, T), T, 0, stream>>>(features, Ah, (long long)GN * GIN);
  pack_w1t_kernel<<<nblk(128LL * 128, T), T, 0, stream>>>(W1, W1t);
  pack_w2t_kernel<<<nblk(48LL * 128, T), T, 0, stream>>>(W2, W2t);

  // ---- layer 1 ----
  gemm_wmma_kernel<128><<<GNP / 64, 256, 0, stream>>>(Ah, W1t, feat1);
  el_er_kernel<GHID><<<nblk((long long)GN * GHEADS, T), T, 0, stream>>>(feat1, al1, ar1, el1, er1,
                                                                        GN, GHEADS, 128);
  fill_u32_kernel<<<nblk((long long)GN * GHEADS, T), T, 0, stream>>>(emax1, 0u, (long long)GN * GHEADS);
  fill_f32_kernel<<<nblk((long long)GN * GHEADS, T), T, 0, stream>>>(den1, 0.f, (long long)GN * GHEADS);
  fill_f32_kernel<<<nblk((long long)GN * 128, T), T, 0, stream>>>(out1, 0.f, (long long)GN * 128);

  edge_max_kernel  <<<nblk((long long)GE * GHEADS, T), T, 0, stream>>>(esrc, edst, el1, er1, emax1, GE, GHEADS);
  edge_denom_kernel<<<nblk((long long)GE * GHEADS, T), T, 0, stream>>>(esrc, edst, el1, er1, emax1, den1, GE, GHEADS);
  edge_aggr_kernel<GHID><<<nblk((long long)GE * GHEADS, T), T, 0, stream>>>(esrc, edst, el1, er1, emax1, den1,
                                                                            feat1, out1, GE, GHEADS, 128, 128);
  finalize1_kernel<<<nblk((long long)GN * 128, T), T, 0, stream>>>(out1, b1, hh, (long long)GN * 128);

  // ---- layer 2 (1 head, CLS=40 padded to 48 for 16-wide WMMA tiles) ----
  gemm_wmma_kernel<48><<<GNP / 64, 256, 0, stream>>>(hh, W2t, feat2);
  el_er_kernel<GCLS><<<nblk((long long)GN, T), T, 0, stream>>>(feat2, al2, ar2, el2, er2, GN, 1, 48);
  fill_u32_kernel<<<nblk((long long)GN, T), T, 0, stream>>>(emax2, 0u, (long long)GN);
  fill_f32_kernel<<<nblk((long long)GN, T), T, 0, stream>>>(den2, 0.f, (long long)GN);
  fill_f32_kernel<<<nblk((long long)GN * GCLS, T), T, 0, stream>>>(out, 0.f, (long long)GN * GCLS);

  edge_max_kernel  <<<nblk((long long)GE, T), T, 0, stream>>>(esrc, edst, el2, er2, emax2, GE, 1);
  edge_denom_kernel<<<nblk((long long)GE, T), T, 0, stream>>>(esrc, edst, el2, er2, emax2, den2, GE, 1);
  edge_aggr_kernel<GCLS><<<nblk((long long)GE, T), T, 0, stream>>>(esrc, edst, el2, er2, emax2, den2,
                                                                   feat2, out, GE, 1, 48, GCLS);
  add_bias_kernel<<<nblk((long long)GN * GCLS, T), T, 0, stream>>>(out, b2, (long long)GN * GCLS, GCLS);
}
